// CQT2010v2_29815662969010
// MI455X (gfx1250) — matrix operand order repaired and checked by reading, not compiled, not run on hardware
//
#include <hip/hip_runtime.h>

// CQT2010v2 for MI455X (gfx1250, wave32, WMMA).
//
// Pipeline (all on one f16 signal pyramid in d_ws):
//   0) cvt16_k: x (f32) -> pyramid level 0 (f16), one pass.
//   1) downsample16_k x6: 256-tap half-band, stride 2, f32 accumulate,
//      f16 in/out -> pyramid levels 1..6.
//   2) cqt_wmma_k: per (batch, octave) the strided CQT conv is a GEMM
//      C[4097 x 24] = A[4097 x 256] x B[256 x 24] in 16x16 WMMA tiles
//      (v_wmma_f32_16x16x32_f16, 8 K-steps). Interior tiles: A fragment is
//      2x global_load_b128 of f16 runs -- no conversion VALU in the hot loop.
//      Edge tiles pay branchless reflect indexing (wave-uniform branch keeps
//      EXEC all-ones for WMMA). Columns interleave real/imag per bin so one
//      __shfl_xor(.,1) fuses sqrt(re^2+im^2+1e-8)*sqrt(len).
//
// Roofline: ~28 GFLOP vs ~230MB moved -> compute-bound; f16 WMMA w/ f32
// accumulate is the right pipe; decimation (N=1 channel) stays on VALU.

typedef __attribute__((ext_vector_type(16))) _Float16 v16h;
typedef __attribute__((ext_vector_type(8)))  _Float16 v8h;
typedef __attribute__((ext_vector_type(8)))  float    v8f;

#define LCONST   (1 << 21)      // samples per signal
#define BATCH    16
#define KW       256            // CQT kernel width
#define PADK     128            // reflect pad = KW/2
#define NOCT     7
#define NBINS    84
#define TFR      4097           // frames per octave = L/512 + 1
#define NTILES   257            // ceil(4097/16)
#define WPB      8              // waves per block (256 threads)
#define GRPS     33             // ceil(NTILES/WPB)
#define PBAT     (2 * LCONST - (LCONST >> 6))  // halves per batch (lvl0 + pyramid)
#define BROW     264            // padded LDS row (256 + 8 halves) -> conflict-free

// half-offset of pyramid level d within one batch (d = 0..6)
__device__ __forceinline__ int off16(int d) {
    return (d == 0) ? 0 : (2 * LCONST - (LCONST >> (d - 1)));
}

// ---------------------------------------------------------------------------
// f32 -> f16 conversion of level 0 (8 elements/thread, b128 in / b128 out).
// ---------------------------------------------------------------------------
__global__ __launch_bounds__(256) void cvt16_k(
    const float* __restrict__ x, _Float16* __restrict__ pyr)
{
    const int b = blockIdx.y;
    const size_t i = ((size_t)blockIdx.x * 256 + threadIdx.x) * 8;
    const float* src = x + (size_t)b * LCONST + i;
    const float4 q0 = *(const float4*)(src);
    const float4 q1 = *(const float4*)(src + 4);
    v8h h;
    h[0] = (_Float16)q0.x; h[1] = (_Float16)q0.y;
    h[2] = (_Float16)q0.z; h[3] = (_Float16)q0.w;
    h[4] = (_Float16)q1.x; h[5] = (_Float16)q1.y;
    h[6] = (_Float16)q1.z; h[7] = (_Float16)q1.w;
    *(v8h*)(pyr + (size_t)b * PBAT + i) = h;
}

// ---------------------------------------------------------------------------
// Half-band decimator on f16 pyramid: out[t] = sum_k lp[k]*in[2t+k-127].
// f32 accumulate; LDS-staged taps + input tile.
// ---------------------------------------------------------------------------
__global__ __launch_bounds__(256) void downsample16_k(
    const float* __restrict__ lp, _Float16* __restrict__ pyr, int d)
{
    __shared__ float lps[256];
    __shared__ float ins[768];
    const int tid  = threadIdx.x;
    const int b    = blockIdx.y;
    const int Lin  = LCONST >> (d - 1);
    const int Lout = LCONST >> d;
    const _Float16* in  = pyr + (size_t)b * PBAT + off16(d - 1);
    _Float16*       out = pyr + (size_t)b * PBAT + off16(d);

    const int t0 = blockIdx.x * 256;
    lps[tid] = lp[tid];
    const int base = 2 * t0 - 127;
    for (int i = tid; i < 768; i += 256) {
        int j = base + i;
        ins[i] = (j >= 0 && j < Lin) ? (float)in[j] : 0.0f;
    }
    __syncthreads();

    float acc = 0.0f;
#pragma unroll 8
    for (int k = 0; k < 256; ++k)
        acc = fmaf(lps[k], ins[2 * tid + k], acc);

    const int t = t0 + tid;
    if (t < Lout) out[t] = (_Float16)acc;
}

// ---------------------------------------------------------------------------
// CQT octave conv as f16 WMMA GEMM + fused complex magnitude.
// Column mapping: col c in [0,32): bin = c>>1, part = c&1 (0=real, 1=-imag).
// ---------------------------------------------------------------------------
__global__ __launch_bounds__(256) void cqt_wmma_k(
    const float* __restrict__ kre, const float* __restrict__ kim,
    const float* __restrict__ lengths, const _Float16* __restrict__ pyr,
    float* __restrict__ out)
{
    __shared__ _Float16 Bs[32 * BROW];   // 32 cols x 256 taps (padded), f16

    const int tid = threadIdx.x;
    const int blk = blockIdx.x;
    const int g   = blk % GRPS;
    const int d   = (blk / GRPS) % NOCT;
    const int b   = blk / (GRPS * NOCT);

    // Stage kernel matrix (real / negated imag) into LDS as f16.
    for (int i = tid; i < 32 * 256; i += 256) {
        const int c = i >> 8, k = i & 255;
        const int kidx = c >> 1, part = c & 1;
        float v = 0.0f;
        if (kidx < 12) v = part ? -kim[kidx * 256 + k] : kre[kidx * 256 + k];
        Bs[c * BROW + k] = (_Float16)v;
    }
    __syncthreads();

    const int wave = tid >> 5, lane = tid & 31;
    const int tile = g * WPB + wave;
    if (tile >= NTILES) return;

    const int half  = lane >> 4;     // K half-wave select
    const int lan15 = lane & 15;     // A: row M   | B/C: column N
    const int hop   = 512 >> d;
    const int Ln    = LCONST >> d;
    const _Float16* sig = pyr + (size_t)b * PBAT + off16(d);

    const int m0     = tile * 16;
    const int tfr    = m0 + lan15;                    // frame for this A row
    const int abase0 = tfr * hop - PADK + half * 8;   // window base (pre-reflect)
    const int reflim = 2 * Ln - 2;

    const _Float16* brow = &Bs[lan15 * BROW];         // B column for this lane

    v8f acc0 = {};   // columns  0..15 (bins 0..7)
    v8f acc1 = {};   // columns 16..31 (bins 8..15; 12..15 zero-padded)

    // Wave-uniform interior test: every lane's K-window inside [0, Ln).
    //   min index = m0*hop - 128 ; max index = (m0+15)*hop + 127
    const bool interior = (m0 * hop >= PADK) && ((m0 + 15) * hop + 127 < Ln);

    if (interior) {
        // Fast path: each A fragment = two contiguous 8-half runs, 16B-aligned
        // (abase0 is a multiple of 8 since hop >= 8) -> 2x global_load_b128,
        // zero conversion VALU.
#pragma unroll 2
        for (int ks = 0; ks < 8; ++ks) {
            const _Float16* p = sig + (abase0 + ks * 32);
            const v8h lo = *(const v8h*)(p);        // elems 0..7
            const v8h hi = *(const v8h*)(p + 16);   // elems 8..15
            v16h a;
#pragma unroll
            for (int i = 0; i < 8; ++i) { a[i] = lo[i]; a[8 + i] = hi[i]; }

            const _Float16* bp0 = brow + ks * 32 + half * 8;
            const _Float16* bp1 = bp0 + 16 * BROW;
            v16h b0, b1;
#pragma unroll
            for (int i = 0; i < 8; ++i) { b0[i] = bp0[i]; b0[8 + i] = bp0[16 + i]; b1[i] = bp1[i]; b1[8 + i] = bp1[16 + i]; }

            acc0 = __builtin_amdgcn_wmma_f32_16x16x32_f16(
                false, a, false, b0, (short)0, acc0, false, false);
            acc1 = __builtin_amdgcn_wmma_f32_16x16x32_f16(
                false, a, false, b1, (short)0, acc1, false, false);
        }
    } else {
        // Edge path: per-element branchless reflect + clamp (f16 loads).
        for (int ks = 0; ks < 8; ++ks) {
            const int ab = abase0 + ks * 32;
            v16h a;
#pragma unroll
            for (int i = 0; i < 8; ++i) {
                int j = ab + i;
                j = j < 0 ? -j : j;           // reflect left
                j = min(j, reflim - j);       // reflect right
                j = max(0, min(j, Ln - 1));   // safety clamp
                a[i] = sig[j];
            }
#pragma unroll
            for (int i = 0; i < 8; ++i) {
                int j = ab + 16 + i;
                j = j < 0 ? -j : j;
                j = min(j, reflim - j);
                j = max(0, min(j, Ln - 1));
                a[8 + i] = sig[j];
            }

            const _Float16* bp0 = brow + ks * 32 + half * 8;
            const _Float16* bp1 = bp0 + 16 * BROW;
            v16h b0, b1;
#pragma unroll
            for (int i = 0; i < 8; ++i) { b0[i] = bp0[i]; b0[8 + i] = bp0[16 + i]; b1[i] = bp1[i]; b1[8 + i] = bp1[16 + i]; }

            acc0 = __builtin_amdgcn_wmma_f32_16x16x32_f16(
                false, a, false, b0, (short)0, acc0, false, false);
            acc1 = __builtin_amdgcn_wmma_f32_16x16x32_f16(
                false, a, false, b1, (short)0, acc1, false, false);
        }
    }

    // C/D layout: lane l holds column n=l&15, rows m = r + 8*(l>>4).
    // Adjacent lanes (xor 1) hold the real/imag partner column of same bin.
    const int  n      = lan15;
    const bool isReal = (n & 1) == 0;
    const int  bin0   = n >> 1;                    // nt=0: bins 0..7, valid
    const int  bin1   = 8 + bin0;                  // nt=1: bins 8..15
    const int  bf0    = (6 - d) * 12 + bin0;
    const int  bf1c   = min((6 - d) * 12 + bin1, NBINS - 1);  // clamp padded bins
    const float sl0   = sqrtf(lengths[bf0]);
    const float sl1   = sqrtf(lengths[bf1c]);

#pragma unroll
    for (int r = 0; r < 8; ++r) {
        const int t = m0 + half * 8 + r;
        const float v0 = acc0[r];
        const float o0 = __shfl_xor(v0, 1, 32);
        const float v1 = acc1[r];
        const float o1 = __shfl_xor(v1, 1, 32);
        if (isReal && t < TFR) {
            out[((size_t)(b * NBINS + bf0)) * TFR + t] =
                sqrtf(v0 * v0 + o0 * o0 + 1e-8f) * sl0;
            if (bin1 < 12) {
                out[((size_t)(b * NBINS + (6 - d) * 12 + bin1)) * TFR + t] =
                    sqrtf(v1 * v1 + o1 * o1 + 1e-8f) * sl1;
            }
        }
    }
}

// ---------------------------------------------------------------------------
extern "C" void kernel_launch(void* const* d_in, const int* in_sizes, int n_in,
                              void* d_out, int out_size, void* d_ws, size_t ws_size,
                              hipStream_t stream) {
    const float* x   = (const float*)d_in[0];   // (16,1,2^21)
    const float* kre = (const float*)d_in[1];   // (12,1,256)
    const float* kim = (const float*)d_in[2];   // (12,1,256)
    const float* lp  = (const float*)d_in[3];   // (1,1,256)
    const float* len = (const float*)d_in[4];   // (84,)
    _Float16* pyr = (_Float16*)d_ws;            // f16 pyramid, ~127 MB
    float* out = (float*)d_out;                 // (16,84,4097)

    // Level 0: f32 -> f16 (one pass over the input).
    cvt16_k<<<dim3(LCONST / (256 * 8), BATCH), 256, 0, stream>>>(x, pyr);

    // Decimation chain: level d depends on level d-1 (stream-serialized).
    for (int d = 1; d <= 6; ++d) {
        const int Lout = LCONST >> d;
        dim3 grid((Lout + 255) / 256, BATCH);
        downsample16_k<<<grid, 256, 0, stream>>>(lp, pyr, d);
    }

    // One WMMA launch covers all (batch, octave, frame-tile) work.
    cqt_wmma_k<<<dim3(GRPS * NOCT * BATCH), 256, 0, stream>>>(
        kre, kim, len, pyr, out);
}
